// SelfAttention_39599598469410
// MI455X (gfx1250) — matrix operand order repaired and checked by reading, not compiled, not run on hardware
//
#include <hip/hip_runtime.h>
#include <math.h>

// ---- problem constants -----------------------------------------------------
#define BB 8
#define TT 1024
#define CC 1024
#define HH 16
#define HD 64
#define NEGBIG (-1000000000.0f)

typedef __bf16 bf16;
typedef __bf16 v16bf __attribute__((ext_vector_type(16)));
typedef __bf16 bf16x8 __attribute__((ext_vector_type(8)));
typedef __bf16 bf16x4 __attribute__((ext_vector_type(4)));
typedef float  v8f   __attribute__((ext_vector_type(8)));

// ---- WMMA wrapper ----------------------------------------------------------
__device__ __forceinline__ v8f wmma_bf16(v16bf a, v16bf b, v8f c) {
  // (neg_a, A, neg_b, B, c_mod, C, reuse_a, reuse_b)
  return __builtin_amdgcn_wmma_f32_16x16x32_bf16(false, a, false, b, (short)0, c,
                                                 false, false);
}

// A operand, 16x32 (MxK), from row-major src (row stride ld elements).
// lane<16: K=0..15; lane>=16: K=16..31 (contiguous); M = lane&15.
__device__ __forceinline__ v16bf load_fragA(const bf16* __restrict__ base, int ld,
                                            int row0, int k0, int lane) {
  int r  = row0 + (lane & 15);
  int kb = k0 + ((lane >> 4) << 4);
  const bf16* p = base + (size_t)r * ld + kb;
  union { v16bf v; bf16x8 h[2]; } u;
  u.h[0] = *(const bf16x8*)(p);
  u.h[1] = *(const bf16x8*)(p + 8);
  return u.v;
}

// B operand, 32x16 (KxN), column n = lane&15 read from row-major src rows
// (src[n, k]): lane<16 holds K=0..15, lane>=16 holds K=16..31 (contiguous).
__device__ __forceinline__ v16bf load_fragB(const bf16* __restrict__ base, int ld,
                                            int col0, int k0, int lane) {
  int n  = col0 + (lane & 15);
  int kb = k0 + ((lane >> 4) << 4);
  const bf16* p = base + (size_t)n * ld + kb;
  union { v16bf v; bf16x8 h[2]; } u;
  u.h[0] = *(const bf16x8*)(p);
  u.h[1] = *(const bf16x8*)(p + 8);
  return u.v;
}

// ---- fp32 -> bf16 convert --------------------------------------------------
__global__ __launch_bounds__(256)
void cvt_f32_bf16(const float* __restrict__ in, bf16* __restrict__ out, int n) {
  int i = (blockIdx.x * 256 + threadIdx.x) * 4;
  if (i + 3 < n) {
    float4 f = *(const float4*)(in + i);
    bf16x4 w;
    w[0] = (bf16)f.x; w[1] = (bf16)f.y; w[2] = (bf16)f.z; w[3] = (bf16)f.w;
    *(bf16x4*)(out + i) = w;
  }
}

// ---- generic WMMA GEMM: out[m,n] = sum_k A[m,k]*W[n,k] + bias[n] ----------
// Wave tile 64x32 (4 M-tiles x 2 N-tiles): 8 WMMAs per 6 fragment loads,
// keeping the XDL path compute-bound vs the per-SIMD load path.
// MODE 0: bf16 out as [B,H,T,HD]   (Q / K)
// MODE 1: bf16 out as [B,H,HD,T]   (V transposed)
// MODE 2: fp32 out row-major [M,C] (projection)
template <int MODE>
__global__ __launch_bounds__(256)
void gemm_wmma_kernel(const bf16* __restrict__ A, const bf16* __restrict__ W,
                      const float* __restrict__ bias, void* __restrict__ Out) {
  const int K = CC;
  int lane = threadIdx.x & 31;
  int wave = threadIdx.x >> 5;
  int wm = wave >> 2, wn = wave & 3;       // 2x4 waves -> WG tile 128x128
  int bm = blockIdx.x & 63;                // 8192/128 = 64 M blocks
  int bn = blockIdx.x >> 6;                // 1024/128 = 8  N blocks
  int m0 = bm * 128 + wm * 64;
  int n0 = bn * 128 + wn * 32;

  v8f acc[4][2];
#pragma unroll
  for (int mt = 0; mt < 4; ++mt)
#pragma unroll
    for (int nt = 0; nt < 2; ++nt)
#pragma unroll
      for (int i = 0; i < 8; ++i) acc[mt][nt][i] = 0.0f;

  for (int k0 = 0; k0 < K; k0 += 32) {
    v16bf b0 = load_fragB(W, K, n0,      k0, lane);
    v16bf b1 = load_fragB(W, K, n0 + 16, k0, lane);
#pragma unroll
    for (int mt = 0; mt < 4; ++mt) {
      v16bf a = load_fragA(A, K, m0 + mt * 16, k0, lane);
      acc[mt][0] = wmma_bf16(a, b0, acc[mt][0]);
      acc[mt][1] = wmma_bf16(a, b1, acc[mt][1]);
    }
  }

  int halfsel = (lane >> 4) << 3;          // C/D: lanes>=16 hold M = i+8
#pragma unroll
  for (int nt = 0; nt < 2; ++nt) {
    int n = n0 + nt * 16 + (lane & 15);
    float bv = bias[n];
    int h = n >> 6, d = n & 63;
#pragma unroll
    for (int mt = 0; mt < 4; ++mt) {
#pragma unroll
      for (int i = 0; i < 8; ++i) {
        float v = acc[mt][nt][i] + bv;
        int m = m0 + mt * 16 + halfsel + i;
        int b = m >> 10, t = m & 1023;
        if (MODE == 0) {
          ((bf16*)Out)[(((size_t)(b * HH + h) * TT + t) << 6) + d] = (bf16)v;
        } else if (MODE == 1) {
          ((bf16*)Out)[(((size_t)(b * HH + h) * HD + d) << 10) + t] = (bf16)v;
        } else {
          ((float*)Out)[(size_t)m * CC + n] = v;
        }
      }
    }
  }
}

// ---- flash attention: S^T = K.Q^T trick, online softmax, O^T = V^T.P^T ----
__global__ __launch_bounds__(256)
void attn_wmma_kernel(const bf16* __restrict__ Q, const bf16* __restrict__ Km,
                      const bf16* __restrict__ Vt, const int* __restrict__ mask,
                      bf16* __restrict__ Y) {
  __shared__ float sbias[TT];
  int b  = blockIdx.x >> 7;                 // 8 batches
  int rem = blockIdx.x & 127;
  int h  = rem >> 3;                        // 16 heads
  int qc = rem & 7;                         // 8 q-chunks of 128 rows

  // mask bias for this batch (additive -1e9 on masked keys)
  for (int idx = threadIdx.x; idx < TT; idx += 256)
    sbias[idx] = mask[b * TT + idx] ? 0.0f : NEGBIG;
  __syncthreads();

  int lane = threadIdx.x & 31;
  int wave = threadIdx.x >> 5;
  int q0 = (qc * 8 + wave) * 16;            // 16 queries per wave
  int halfsel = (lane >> 4) << 3;
  const float SC = 0.125f;                  // 1/sqrt(64)

  const bf16* Qb  = Q  + (size_t)(b * HH + h) * TT * HD;   // [T,64]
  const bf16* Kb  = Km + (size_t)(b * HH + h) * TT * HD;   // [T,64]
  const bf16* Vtb = Vt + (size_t)(b * HH + h) * HD * TT;   // [64,T]

  // Q^T as B operand (N = q column, K = d), held for the whole loop
  v16bf bq0 = load_fragB(Qb, HD, q0, 0,  lane);
  v16bf bq1 = load_fragB(Qb, HD, q0, 32, lane);

  float mrun = -3.0e38f, lrun = 0.0f;
  v8f o[4];
#pragma unroll
  for (int t = 0; t < 4; ++t)
#pragma unroll
    for (int i = 0; i < 8; ++i) o[t][i] = 0.0f;

  for (int j0 = 0; j0 < TT; j0 += 32) {
    // S^T tiles: M = key rows, N = q columns, K = d (two k=32 steps)
    v16bf ka00 = load_fragA(Kb, HD, j0,      0,  lane);
    v16bf ka01 = load_fragA(Kb, HD, j0,      32, lane);
    v16bf ka10 = load_fragA(Kb, HD, j0 + 16, 0,  lane);
    v16bf ka11 = load_fragA(Kb, HD, j0 + 16, 32, lane);
    v8f st0, st1;
#pragma unroll
    for (int i = 0; i < 8; ++i) { st0[i] = 0.0f; st1[i] = 0.0f; }
    st0 = wmma_bf16(ka00, bq0, st0);
    st0 = wmma_bf16(ka01, bq1, st0);
    st1 = wmma_bf16(ka10, bq0, st1);
    st1 = wmma_bf16(ka11, bq1, st1);

    // scale + key mask (key index = j0 [+16] + halfsel + i)
    float mb0[8], mb1[8];
    *(float4*)&mb0[0] = *(const float4*)&sbias[j0 + halfsel];
    *(float4*)&mb0[4] = *(const float4*)&sbias[j0 + halfsel + 4];
    *(float4*)&mb1[0] = *(const float4*)&sbias[j0 + 16 + halfsel];
    *(float4*)&mb1[4] = *(const float4*)&sbias[j0 + 16 + halfsel + 4];
    float s0[8], s1[8];
#pragma unroll
    for (int i = 0; i < 8; ++i) {
      s0[i] = st0[i] * SC + mb0[i];
      s1[i] = st1[i] * SC + mb1[i];
    }

    // running max: in-lane over 16 keys, then merge lane halves (same q col)
    float tmax = s0[0];
#pragma unroll
    for (int i = 1; i < 8; ++i) tmax = fmaxf(tmax, s0[i]);
#pragma unroll
    for (int i = 0; i < 8; ++i) tmax = fmaxf(tmax, s1[i]);
    tmax = fmaxf(tmax, __shfl_xor(tmax, 16));
    float mnew  = fmaxf(mrun, tmax);
    float alpha = __expf(mrun - mnew);
    mrun = mnew;

    float p0[8], p1[8], psum = 0.0f;
#pragma unroll
    for (int i = 0; i < 8; ++i) { p0[i] = __expf(s0[i] - mnew); psum += p0[i]; }
#pragma unroll
    for (int i = 0; i < 8; ++i) { p1[i] = __expf(s1[i] - mnew); psum += p1[i]; }
    lrun = lrun * alpha + psum;

    // Repack P^T (C/D layout) into B-operand layout for K=32:
    // lane<16 needs keys {0..15}: own j0..7 + partner's j8..15
    // lane>=16 needs keys {16..31}: partner's j16..23 + own j24..31
    union PB { bf16x8 v; unsigned u[4]; };
    PB lo, hi;
#pragma unroll
    for (int i = 0; i < 8; ++i) { lo.v[i] = (bf16)p0[i]; hi.v[i] = (bf16)p1[i]; }
    PB send; send.v = (lane < 16) ? hi.v : lo.v;
    PB recv;
#pragma unroll
    for (int r = 0; r < 4; ++r)
      recv.u[r] = (unsigned)__shfl_xor((int)send.u[r], 16);
    union { v16bf v; bf16x8 h[2]; } pfrag;
    pfrag.h[0] = (lane < 16) ? lo.v : recv.v;
    pfrag.h[1] = (lane < 16) ? recv.v : hi.v;

    // O^T accumulate: A = V^T d-tiles (M=d, K=key), rescale C by alpha
#pragma unroll
    for (int t = 0; t < 4; ++t) {
      v16bf va = load_fragA(Vtb, TT, t * 16, j0, lane);
      v8f ca;
#pragma unroll
      for (int i = 0; i < 8; ++i) ca[i] = o[t][i] * alpha;
      o[t] = wmma_bf16(va, pfrag.v, ca);
    }
  }

  float ltot = lrun + __shfl_xor(lrun, 16);
  float inv  = 1.0f / ltot;

  // O^T element (d = t*16 + halfsel + i, q = q0 + lane&15) -> y[b, q, h*64+d]
  int q = q0 + (lane & 15);
  bf16* yrow = Y + ((size_t)(b * TT + q)) * CC + h * HD;
#pragma unroll
  for (int t = 0; t < 4; ++t) {
    bf16x8 w;
#pragma unroll
    for (int i = 0; i < 8; ++i) w[i] = (bf16)(o[t][i] * inv);
    *(bf16x8*)(yrow + t * 16 + halfsel) = w;
  }
}

// ---- residual + LayerNorm (eps=1e-12), fp32 out ---------------------------
__global__ __launch_bounds__(256)
void resid_ln_kernel(const float* __restrict__ P, const float* __restrict__ x,
                     const float* __restrict__ lnw, const float* __restrict__ lnb,
                     float* __restrict__ out) {
  __shared__ float reds[8], redq[8];
  int row = blockIdx.x;
  int tid = threadIdx.x;
  int lane = tid & 31, wave = tid >> 5;
  int c = tid * 4;
  size_t base = (size_t)row * CC + c;
  float4 pv = *(const float4*)(P + base);
  float4 xv = *(const float4*)(x + base);
  float z0 = pv.x + xv.x, z1 = pv.y + xv.y, z2 = pv.z + xv.z, z3 = pv.w + xv.w;
  float s  = z0 + z1 + z2 + z3;
  float sq = z0 * z0 + z1 * z1 + z2 * z2 + z3 * z3;
#pragma unroll
  for (int off = 16; off >= 1; off >>= 1) {
    s  += __shfl_xor(s, off);
    sq += __shfl_xor(sq, off);
  }
  if (lane == 0) { reds[wave] = s; redq[wave] = sq; }
  __syncthreads();
  float ts = 0.0f, tq = 0.0f;
#pragma unroll
  for (int w = 0; w < 8; ++w) { ts += reds[w]; tq += redq[w]; }
  float mu  = ts * (1.0f / CC);
  float var = tq * (1.0f / CC) - mu * mu;
  float r   = rsqrtf(var + 1e-12f);
  float4 wv = *(const float4*)(lnw + c);
  float4 bv = *(const float4*)(lnb + c);
  float4 ov;
  ov.x = (z0 - mu) * r * wv.x + bv.x;
  ov.y = (z1 - mu) * r * wv.y + bv.y;
  ov.z = (z2 - mu) * r * wv.z + bv.z;
  ov.w = (z3 - mu) * r * wv.w + bv.w;
  *(float4*)(out + base) = ov;
}

// ---- launcher --------------------------------------------------------------
extern "C" void kernel_launch(void* const* d_in, const int* in_sizes, int n_in,
                              void* d_out, int out_size, void* d_ws, size_t ws_size,
                              hipStream_t stream) {
  const float* x    = (const float*)d_in[0];
  const int*   mask = (const int*)d_in[1];
  const float* Wq   = (const float*)d_in[2];
  const float* bq   = (const float*)d_in[3];
  const float* Wk   = (const float*)d_in[4];
  const float* bk   = (const float*)d_in[5];
  const float* Wv   = (const float*)d_in[6];
  const float* bv   = (const float*)d_in[7];
  const float* Wp   = (const float*)d_in[8];
  const float* bp   = (const float*)d_in[9];
  const float* lnw  = (const float*)d_in[10];
  const float* lnb  = (const float*)d_in[11];
  float* out = (float*)d_out;

  char* ws = (char*)d_ws;
  const size_t NX = (size_t)BB * TT * CC;          // 8,388,608
  const size_t NW = (size_t)CC * CC;               // 1,048,576
  bf16* xb  = (bf16*)(ws);                         // 16 MB
  bf16* wqb = (bf16*)(ws + 16u * 1024 * 1024);
  bf16* wkb = (bf16*)(ws + 18u * 1024 * 1024);
  bf16* wvb = (bf16*)(ws + 20u * 1024 * 1024);
  bf16* wpb = (bf16*)(ws + 22u * 1024 * 1024);
  bf16* Qb  = (bf16*)(ws + 24u * 1024 * 1024);     // [B,H,T,64] 16 MB
  bf16* Kb  = (bf16*)(ws + 40u * 1024 * 1024);     // [B,H,T,64] 16 MB
  bf16* Vtb = (bf16*)(ws + 56u * 1024 * 1024);     // [B,H,64,T] 16 MB
  bf16* Yb  = (bf16*)(ws + 72u * 1024 * 1024);     // [B,T,C]    16 MB
  float* Pf = (float*)(ws + 88u * 1024 * 1024);    // [B*T,C] fp32 32 MB

  // 1) convert to bf16
  cvt_f32_bf16<<<(int)(NX / 1024), 256, 0, stream>>>(x, xb, (int)NX);
  cvt_f32_bf16<<<(int)(NW / 1024), 256, 0, stream>>>(Wq, wqb, (int)NW);
  cvt_f32_bf16<<<(int)(NW / 1024), 256, 0, stream>>>(Wk, wkb, (int)NW);
  cvt_f32_bf16<<<(int)(NW / 1024), 256, 0, stream>>>(Wv, wvb, (int)NW);
  cvt_f32_bf16<<<(int)(NW / 1024), 256, 0, stream>>>(Wp, wpb, (int)NW);

  // 2) Q/K/V projections (WMMA GEMMs), grid = 64 M-blocks x 8 N-blocks
  gemm_wmma_kernel<0><<<512, 256, 0, stream>>>(xb, wqb, bq, (void*)Qb);
  gemm_wmma_kernel<0><<<512, 256, 0, stream>>>(xb, wkb, bk, (void*)Kb);
  gemm_wmma_kernel<1><<<512, 256, 0, stream>>>(xb, wvb, bv, (void*)Vtb);

  // 3) flash attention (B*H*8 workgroups, 8 waves each, 16 queries/wave)
  attn_wmma_kernel<<<BB * HH * 8, 256, 0, stream>>>(Qb, Kb, Vtb, mask, Yb);

  // 4) output projection (fp32 accum written to workspace)
  gemm_wmma_kernel<2><<<512, 256, 0, stream>>>(Yb, wpb, bp, (void*)Pf);

  // 5) residual + LayerNorm -> d_out
  resid_ln_kernel<<<BB * TT, 256, 0, stream>>>(Pf, x, lnw, lnb, out);
}
